// MultiHeadAttention_3753801417323
// MI455X (gfx1250) — matrix operand order repaired and checked by reading, not compiled, not run on hardware
//
#include <hip/hip_runtime.h>
#include <hip/hip_bf16.h>

// ---------------------------------------------------------------------------
// GQA attention block for MI455X (gfx1250, wave32, WMMA + TDM).
// f32->bf16 converts (weights transposed) -> WMMA GEMMs with TDM-staged,
// double-buffered LDS tiles -> RoPE (scale folded into Q) -> flash-style WMMA
// attention (online softmax, double-buffered K/V chunks) -> WMMA GEMM -> f32.
// ---------------------------------------------------------------------------

#define B_   2
#define S_   1024
#define HID_ 2048
#define NH_  32
#define NKV_ 8
#define HD_  64
#define SCALE_ 0.125f

typedef __attribute__((ext_vector_type(16))) __bf16 v16bf;
typedef __attribute__((ext_vector_type(8)))  __bf16 v8bf;
typedef __attribute__((ext_vector_type(4)))  __bf16 v4bf;
typedef __attribute__((ext_vector_type(8)))  float  v8f;
typedef __attribute__((ext_vector_type(4)))  float  f32x4;
typedef __attribute__((ext_vector_type(4)))  unsigned int u32x4;
typedef __attribute__((ext_vector_type(8)))  unsigned int u32x8;

static __device__ inline v8f wmma_bf16(v16bf a, v16bf b, v8f c) {
  return __builtin_amdgcn_wmma_f32_16x16x32_bf16(false, a, false, b, (short)0, c, false, false);
}

// Fragment = two contiguous 8-elem runs (ISA 7.12.2 pair order == memory order)
static __device__ inline v16bf ldfrag(const __bf16* p0, const __bf16* p1) {
  union { v16bf v; v8bf h[2]; } u;
  u.h[0] = *(const v8bf*)p0;
  u.h[1] = *(const v8bf*)p1;
  return u.v;
}

// Generic->LDS byte offset: low 32 bits of the generic address are the LDS
// offset on gfx1250 (ISA 10.2 aperture mapping).
static __device__ inline unsigned lds_off(const void* p) {
  return (unsigned)(unsigned long long)p;
}

// ---------------------------------------------------------------------------
// Tensor Data Mover: async DMA of a 2D bf16 tile (tile1 rows x tile0 cols)
// from a row-major tensor (row stride = stride0 elems) into LDS, rows packed
// contiguously. Descriptor per ISA cdna5 §8.3/8.4 (groups 0 and 1; 2D form,
// VADDR2/3 = NULL). Issued once per wave; tracked by TENSORcnt.
static __device__ inline void tdm_load_2d_bf16(unsigned ldsoff, const void* gptr,
                                               unsigned tdim0, unsigned tdim1,
                                               unsigned tile0, unsigned tile1,
                                               unsigned stride0) {
  unsigned long long ga = (unsigned long long)gptr;
  u32x4 g0;
  g0[0] = 1u;                                   // count=1 (valid user D#)
  g0[1] = ldsoff;                               // lds_addr (bytes)
  g0[2] = (unsigned)ga;                         // global_addr[31:0]
  g0[3] = ((unsigned)(ga >> 32) & 0x01FFFFFFu)  // global_addr[56:32]
          | 0x80000000u;                        // type=2 ("image")
  u32x8 g1;
  g1[0] = (1u << 16);                           // data_size=1 (2 bytes)
  g1[1] = (tdim0 & 0xFFFFu) << 16;              // tensor_dim0[15:0]
  g1[2] = (tdim0 >> 16) | ((tdim1 & 0xFFFFu) << 16);  // dim0[31:16], dim1[15:0]
  g1[3] = (tdim1 >> 16) | (tile0 << 16);        // dim1[31:16], tile_dim0
  g1[4] = tile1;                                // tile_dim1 (tile_dim2 = 0)
  g1[5] = stride0;                              // tensor_dim0_stride[31:0]
  g1[6] = 0;                                    // stride0[47:32], stride1[15:0]
  g1[7] = 0;
  asm volatile("tensor_load_to_lds %0, %1" :: "s"(g0), "s"(g1) : "memory");
}

// ---------------------------------------------------------------------------
// vectorized f32 -> bf16 (n multiple of 4)
__global__ void cvt_f32_bf16_kernel(const float* __restrict__ src,
                                    __bf16* __restrict__ dst, int n4) {
  int i = blockIdx.x * blockDim.x + threadIdx.x;
  if (i >= n4) return;
  f32x4 x = *(const f32x4*)(src + 4 * (size_t)i);
  v4bf y;
  y[0] = (__bf16)x[0]; y[1] = (__bf16)x[1];
  y[2] = (__bf16)x[2]; y[3] = (__bf16)x[3];
  *(v4bf*)(dst + 4 * (size_t)i) = y;
}

// weight convert + transpose: src f32 [K][N] -> dst bf16 [N][K]
__global__ void wt_cvt_kernel(const float* __restrict__ src,
                              __bf16* __restrict__ dst, int N, int K) {
  int i = blockIdx.x * blockDim.x + threadIdx.x;
  if (i >= N * K) return;
  int n = i % N, k = i / N;   // n fastest -> coalesced reads
  dst[(size_t)n * K + k] = (__bf16)src[i];
}

// ---------------------------------------------------------------------------
// C[M,N](f32) = A[M,K](bf16) * W[K,N]; W passed TRANSPOSED as Wt[N][K].
// 8 waves; WG tile 128x128; wave tile 32x64. K staged 64/step via TDM into
// double-buffered LDS; wave 0 drives the DMA, all waves compute.
__global__ __launch_bounds__(256)
void gemm_bf16_wmma_kernel(float* __restrict__ C, const __bf16* __restrict__ A,
                           const __bf16* __restrict__ Wt, int M, int N, int K) {
  __shared__ __attribute__((aligned(16))) __bf16 As[2][128][64];
  __shared__ __attribute__((aligned(16))) __bf16 Bs[2][128][64];

  const int tid  = threadIdx.x;
  const int wave = tid >> 5;
  const int lane = tid & 31;
  const int hl   = lane >> 4;
  const int l16  = lane & 15;

  const int m0 = blockIdx.y * 128;
  const int n0 = blockIdx.x * 128;
  const int wm = (wave & 3) * 32;
  const int wn = (wave >> 2) * 64;

  v8f acc[2][4] = {};

  const int nt = K / 64;
  if (wave == 0) {
    tdm_load_2d_bf16(lds_off(&As[0][0][0]), A + (size_t)m0 * K, K, M, 64, 128, K);
    tdm_load_2d_bf16(lds_off(&Bs[0][0][0]), Wt + (size_t)n0 * K, K, N, 64, 128, K);
    __builtin_amdgcn_s_wait_tensorcnt(0);
  }
  __syncthreads();

  for (int t = 0; t < nt; ++t) {
    const int cur = t & 1;
    if (t + 1 < nt && wave == 0) {
      const int kc = (t + 1) * 64;
      tdm_load_2d_bf16(lds_off(&As[cur ^ 1][0][0]), A + (size_t)m0 * K + kc, K, M, 64, 128, K);
      tdm_load_2d_bf16(lds_off(&Bs[cur ^ 1][0][0]), Wt + (size_t)n0 * K + kc, K, N, 64, 128, K);
    }

#pragma unroll
    for (int k2 = 0; k2 < 2; ++k2) {
      const int ko = k2 * 32;
      v16bf afr[2], bfr[4];
#pragma unroll
      for (int mi = 0; mi < 2; ++mi) {
        const __bf16* arow = &As[cur][wm + mi * 16 + l16][ko];
        afr[mi] = ldfrag(arow + 8 * hl, arow + 16 + 8 * hl);
      }
#pragma unroll
      for (int ni = 0; ni < 4; ++ni) {
        const __bf16* brow = &Bs[cur][wn + ni * 16 + l16][ko];
        bfr[ni] = ldfrag(brow + 8 * hl, brow + 16 + 8 * hl);
      }
#pragma unroll
      for (int mi = 0; mi < 2; ++mi)
#pragma unroll
        for (int ni = 0; ni < 4; ++ni)
          acc[mi][ni] = wmma_bf16(afr[mi], bfr[ni], acc[mi][ni]);
    }

    if (t + 1 < nt) {
      if (wave == 0) __builtin_amdgcn_s_wait_tensorcnt(0);
      __syncthreads();
    }
  }

  // C/D layout: VGPR r -> row r + 8*hl, col l16 (coalesced b32 stores)
#pragma unroll
  for (int mi = 0; mi < 2; ++mi)
#pragma unroll
    for (int ni = 0; ni < 4; ++ni)
#pragma unroll
      for (int r = 0; r < 8; ++r) {
        int row = m0 + wm + mi * 16 + r + 8 * hl;
        int col = n0 + wn + ni * 16 + l16;
        C[(size_t)row * N + col] = acc[mi][ni][r];
      }
}

// ---------------------------------------------------------------------------
// RoPE (+ optional scale): src f32 [B*S][nh*64] -> dst bf16 [B][nh][S][64]
__global__ void rope_kernel(const float* __restrict__ src,
                            const float* __restrict__ cosv,
                            const float* __restrict__ sinv,
                            __bf16* __restrict__ dst, int nh, float scale, int n) {
  int i = blockIdx.x * blockDim.x + threadIdx.x;
  if (i >= n) return;
  int d = i & 31;
  int h = (i >> 5) % nh;
  int s = (i >> 5) / nh % S_;
  int b = i / (32 * nh * S_);
  size_t srow = (size_t)(b * S_ + s) * nh * HD_ + h * HD_;
  float e = src[srow + 2 * d];
  float o = src[srow + 2 * d + 1];
  float c = cosv[(size_t)(b * S_ + s) * 32 + d];
  float sn = sinv[(size_t)(b * S_ + s) * 32 + d];
  size_t drow = ((size_t)(b * nh + h) * S_ + s) * HD_;
  dst[drow + d]      = (__bf16)((e * c - o * sn) * scale);
  dst[drow + d + 32] = (__bf16)((e * sn + o * c) * scale);
}

// V: f32 [B*S][NKV*64] -> bf16 [B][NKV][HD][S] (d-major)
__global__ void v_pack_kernel(const float* __restrict__ src,
                              __bf16* __restrict__ dst, int n) {
  int i = blockIdx.x * blockDim.x + threadIdx.x;
  if (i >= n) return;
  int s = i & (S_ - 1);
  int d = (i >> 10) & 63;
  int h = (i >> 16) & 7;
  int b = i >> 19;
  dst[((size_t)(b * NKV_ + h) * HD_ + d) * S_ + s] =
      (__bf16)src[(size_t)(b * S_ + s) * (NKV_ * HD_) + h * HD_ + d];
}

// ---------------------------------------------------------------------------
// Flash-style attention. Grid (S/128, NH, B); 8 waves x 16 q-rows each.
// kv chunks of 32, double-buffered in LDS; V is [b][h][d][s]; Q pre-scaled.
__global__ __launch_bounds__(256)
void attn_wmma_kernel(const __bf16* __restrict__ Q, const __bf16* __restrict__ K,
                      const __bf16* __restrict__ V, const float* __restrict__ mask,
                      float* __restrict__ Out) {
  __shared__ __attribute__((aligned(16))) __bf16 Klds[2][32][64];  // [kv][d]
  __shared__ __attribute__((aligned(16))) __bf16 Vt[2][64][32];    // [d][kv]
  __shared__ __attribute__((aligned(16))) __bf16 Plds[8][16][32];

  const int tid  = threadIdx.x;
  const int wave = tid >> 5;
  const int lane = tid & 31;
  const int hl   = lane >> 4;
  const int l16  = lane & 15;

  const int b   = blockIdx.z;
  const int h   = blockIdx.y;
  const int kvh = h >> 2;  // GROUPS = 4
  const int q0  = blockIdx.x * 128 + wave * 16;

  const __bf16* Qb = Q + ((size_t)(b * NH_ + h) * S_) * HD_;
  const __bf16* Kb = K + ((size_t)(b * NKV_ + kvh) * S_) * HD_;
  const __bf16* Vb = V + ((size_t)(b * NKV_ + kvh) * HD_) * S_;

  // per-thread staging addresses (one 16B vector each for K and Vt)
  const int kr = tid >> 3, kco = (tid & 7) * 8;
  const int vr = tid >> 2, vco = (tid & 3) * 8;

  // Q fragments for d-chunks 0..31 / 32..63
  v16bf qf[2];
#pragma unroll
  for (int c = 0; c < 2; ++c) {
    const __bf16* qrow = Qb + (size_t)(q0 + l16) * HD_ + c * 32;
    qf[c] = ldfrag(qrow + 8 * hl, qrow + 16 + 8 * hl);
  }

  float mrow[8], lrow[8];
  v8f oacc[4] = {};
#pragma unroll
  for (int r = 0; r < 8; ++r) { mrow[r] = -1e30f; lrow[r] = 0.0f; }

  const int jmax = (blockIdx.x + 1) * 4;  // causal bound for this q-block

  // prologue loads for chunk 0
  u32x4 kreg = *(const u32x4*)&Kb[(size_t)kr * HD_ + kco];
  u32x4 vreg = *(const u32x4*)&Vb[(size_t)vr * S_ + vco];

  for (int j = 0; j < jmax; ++j) {
    const int cur = j & 1;
    *(u32x4*)&Klds[cur][kr][kco] = kreg;
    *(u32x4*)&Vt[cur][vr][vco] = vreg;
    __syncthreads();
    if (j + 1 < jmax) {
      const int kv1 = (j + 1) * 32;
      kreg = *(const u32x4*)&Kb[(size_t)(kv1 + kr) * HD_ + kco];
      vreg = *(const u32x4*)&Vb[(size_t)vr * S_ + kv1 + vco];
    }

    // scores: two 16x16 tiles, d = 2 chunks of 32
    v8f sc[2] = {};
#pragma unroll
    for (int jj = 0; jj < 2; ++jj)
#pragma unroll
      for (int c = 0; c < 2; ++c) {
        const __bf16* krow = &Klds[cur][jj * 16 + l16][c * 32];
        sc[jj] = wmma_bf16(qf[c], ldfrag(krow + 8 * hl, krow + 16 + 8 * hl), sc[jj]);
      }

    // + mask (scale already folded into Q)
    const int kv0 = j * 32;
    float val[2][8];
#pragma unroll
    for (int jj = 0; jj < 2; ++jj)
#pragma unroll
      for (int r = 0; r < 8; ++r) {
        int qrow = q0 + r + 8 * hl;
        int kv = kv0 + jj * 16 + l16;
        val[jj][r] = sc[jj][r] + mask[(size_t)(b * S_ + qrow) * S_ + kv];
      }

    // online softmax (rows live per 16-lane half-group)
#pragma unroll
    for (int r = 0; r < 8; ++r) {
      float vm = fmaxf(val[0][r], val[1][r]);
      vm = fmaxf(vm, __shfl_xor(vm, 1, 32));
      vm = fmaxf(vm, __shfl_xor(vm, 2, 32));
      vm = fmaxf(vm, __shfl_xor(vm, 4, 32));
      vm = fmaxf(vm, __shfl_xor(vm, 8, 32));
      float mn = fmaxf(mrow[r], vm);
      float co = __expf(mrow[r] - mn);
      float p0 = __expf(val[0][r] - mn);
      float p1 = __expf(val[1][r] - mn);
      Plds[wave][r + 8 * hl][l16]      = (__bf16)p0;
      Plds[wave][r + 8 * hl][16 + l16] = (__bf16)p1;
      float rs = p0 + p1;
      rs += __shfl_xor(rs, 1, 32);
      rs += __shfl_xor(rs, 2, 32);
      rs += __shfl_xor(rs, 4, 32);
      rs += __shfl_xor(rs, 8, 32);
      lrow[r] = lrow[r] * co + rs;
      mrow[r] = mn;
#pragma unroll
      for (int t = 0; t < 4; ++t) oacc[t][r] *= co;
    }

    // intra-wave cross-lane LDS RAW on Plds
#if __has_builtin(__builtin_amdgcn_s_wait_dscnt)
    __builtin_amdgcn_s_wait_dscnt(0);
#else
    asm volatile("s_wait_dscnt 0" ::: "memory");
#endif

    // O += P(16x32) @ Vchunk(32x64)
    const __bf16* prow = &Plds[wave][l16][0];
    v16bf pf = ldfrag(prow + 8 * hl, prow + 16 + 8 * hl);
#pragma unroll
    for (int t = 0; t < 4; ++t) {
      const __bf16* vrow = &Vt[cur][t * 16 + l16][0];
      oacc[t] = wmma_bf16(pf, ldfrag(vrow + 8 * hl, vrow + 16 + 8 * hl), oacc[t]);
    }
    __syncthreads();  // all waves done with buf before it is overwritten
  }

  // normalize + write [b][s][h*64+d]
#pragma unroll
  for (int t = 0; t < 4; ++t)
#pragma unroll
    for (int r = 0; r < 8; ++r) {
      int row = q0 + r + 8 * hl;
      int d = t * 16 + l16;
      Out[(size_t)(b * S_ + row) * (NH_ * HD_) + h * HD_ + d] =
          oacc[t][r] / lrow[r];
    }
}

// ---------------------------------------------------------------------------
extern "C" void kernel_launch(void* const* d_in, const int* in_sizes, int n_in,
                              void* d_out, int out_size, void* d_ws, size_t ws_size,
                              hipStream_t stream) {
  (void)in_sizes; (void)n_in; (void)out_size; (void)ws_size;
  const float* hs   = (const float*)d_in[0];
  const float* cosv = (const float*)d_in[1];
  const float* sinv = (const float*)d_in[2];
  const float* mask = (const float*)d_in[3];
  const float* Wq   = (const float*)d_in[4];
  const float* Wk   = (const float*)d_in[5];
  const float* Wv   = (const float*)d_in[6];
  const float* Wo   = (const float*)d_in[7];
  float* out = (float*)d_out;

  const int M = B_ * S_;      // 2048
  const int Nq = NH_ * HD_;   // 2048
  const int Nkv = NKV_ * HD_; // 512

  size_t off = 0;
  auto carve = [&](size_t bytes) {
    void* p = (char*)d_ws + off;
    off += (bytes + 255) & ~(size_t)255;
    return p;
  };
  __bf16* Xbf  = (__bf16*)carve((size_t)M * HID_ * 2);
  __bf16* Wqb  = (__bf16*)carve((size_t)HID_ * Nq * 2);   // [Nq][HID]
  __bf16* Wkb  = (__bf16*)carve((size_t)HID_ * Nkv * 2);  // [Nkv][HID]
  __bf16* Wvb  = (__bf16*)carve((size_t)HID_ * Nkv * 2);  // [Nkv][HID]
  __bf16* Wob  = (__bf16*)carve((size_t)Nq * HID_ * 2);   // [HID][Nq]
  float*  Qf   = (float*)carve((size_t)M * Nq * 4);
  float*  Kf   = (float*)carve((size_t)M * Nkv * 4);
  float*  Vf   = (float*)carve((size_t)M * Nkv * 4);
  __bf16* Qbf  = (__bf16*)carve((size_t)M * Nq * 2);
  __bf16* Kbf  = (__bf16*)carve((size_t)M * Nkv * 2);
  __bf16* Vbf  = (__bf16*)carve((size_t)M * Nkv * 2);
  float*  Attn = (float*)carve((size_t)M * Nq * 4);
  __bf16* Attb = (__bf16*)carve((size_t)M * Nq * 2);

  auto cvt = [&](const float* s, __bf16* d, int n) {
    cvt_f32_bf16_kernel<<<(n / 4 + 255) / 256, 256, 0, stream>>>(s, d, n / 4);
  };
  auto wcvt = [&](const float* s, __bf16* d, int N, int K) {
    int n = N * K;
    wt_cvt_kernel<<<(n + 255) / 256, 256, 0, stream>>>(s, d, N, K);
  };
  cvt(hs, Xbf, M * HID_);
  wcvt(Wq, Wqb, Nq, HID_);
  wcvt(Wk, Wkb, Nkv, HID_);
  wcvt(Wv, Wvb, Nkv, HID_);
  wcvt(Wo, Wob, HID_, Nq);

  // projections
  gemm_bf16_wmma_kernel<<<dim3(Nq / 128, M / 128), 256, 0, stream>>>(Qf, Xbf, Wqb, M, Nq, HID_);
  gemm_bf16_wmma_kernel<<<dim3(Nkv / 128, M / 128), 256, 0, stream>>>(Kf, Xbf, Wkb, M, Nkv, HID_);
  gemm_bf16_wmma_kernel<<<dim3(Nkv / 128, M / 128), 256, 0, stream>>>(Vf, Xbf, Wvb, M, Nkv, HID_);

  // rope + packs (fold softmax scale into Q)
  {
    int nq = B_ * S_ * NH_ * 32;
    rope_kernel<<<(nq + 255) / 256, 256, 0, stream>>>(Qf, cosv, sinv, Qbf, NH_, SCALE_, nq);
    int nk = B_ * S_ * NKV_ * 32;
    rope_kernel<<<(nk + 255) / 256, 256, 0, stream>>>(Kf, cosv, sinv, Kbf, NKV_, 1.0f, nk);
    int nv = B_ * S_ * NKV_ * HD_;
    v_pack_kernel<<<(nv + 255) / 256, 256, 0, stream>>>(Vf, Vbf, nv);
  }

  // attention
  attn_wmma_kernel<<<dim3(S_ / 128, NH_, B_), 256, 0, stream>>>(Qbf, Kbf, Vbf, mask, Attn);

  // output projection
  cvt(Attn, Attb, M * Nq);
  gemm_bf16_wmma_kernel<<<dim3(HID_ / 128, M / 128), 256, 0, stream>>>(out, Attb, Wob, M, HID_, Nq);
}
